// GraphMidpointJointTraining_1726576853099
// MI455X (gfx1250) — matrix operand order repaired and checked by reading, not compiled, not run on hardware
//
#include <hip/hip_runtime.h>
#include <math.h>

// ---- problem constants (match reference) ----
#define NN      100000
#define EE      1600000
#define DH      128
#define EPSC    0.1f

typedef __attribute__((ext_vector_type(16))) _Float16 v16h;
typedef __attribute__((ext_vector_type(8)))  float    v8f;

#define AS_STRIDE 132   // halves per LDS row (padded vs 128 to spread banks)

// ------------------------------------------------------------------
// utility kernels
// ------------------------------------------------------------------
__global__ __launch_bounds__(256) void zero4_kernel(float4* __restrict__ p, int n4) {
    int i = blockIdx.x * 256 + threadIdx.x;
    if (i < n4) p[i] = make_float4(0.f, 0.f, 0.f, 0.f);
}

__global__ __launch_bounds__(256) void cvt_f16_kernel(const float* __restrict__ in,
                                                      _Float16* __restrict__ out, int n) {
    int i = blockIdx.x * 256 + threadIdx.x;
    if (i < n) out[i] = (_Float16)in[i];
}

// ------------------------------------------------------------------
// CSR-by-destination construction (built once per launch, used 16x)
// ------------------------------------------------------------------
__global__ __launch_bounds__(256) void degi_kernel(const int* __restrict__ dst,
                                                   int* __restrict__ degi, int e_cnt) {
    int e = blockIdx.x * 256 + threadIdx.x;
    if (e < e_cnt) atomicAdd(&degi[dst[e]], 1);
}

__global__ __launch_bounds__(256) void dinv_kernel(const int* __restrict__ degi,
                                                   float* __restrict__ dinv, int n) {
    int i = blockIdx.x * 256 + threadIdx.x;
    if (i < n) {
        int d = degi[i];
        dinv[i] = (d > 0) ? rsqrtf((float)d) : 0.f;
    }
}

// per-256-block exclusive scan; block totals to bsum
__global__ __launch_bounds__(256) void scan_block_kernel(const int* __restrict__ in,
                                                         int* __restrict__ out,
                                                         int* __restrict__ bsum, int n) {
    __shared__ int sh[256];
    int tid = threadIdx.x;
    int i   = blockIdx.x * 256 + tid;
    int v   = (i < n) ? in[i] : 0;
    sh[tid] = v;
    for (int ofs = 1; ofs < 256; ofs <<= 1) {
        __syncthreads();
        int t = (tid >= ofs) ? sh[tid - ofs] : 0;
        __syncthreads();
        sh[tid] += t;
    }
    __syncthreads();
    if (i < n) out[i] = sh[tid] - v;           // exclusive
    if (tid == 255) bsum[blockIdx.x] = sh[255]; // block total
}

// single-block in-place exclusive scan of block sums (n <= 512)
__global__ __launch_bounds__(512) void scan_small_kernel(int* __restrict__ data, int n) {
    __shared__ int sh[512];
    int tid = threadIdx.x;
    int v   = (tid < n) ? data[tid] : 0;
    sh[tid] = v;
    for (int ofs = 1; ofs < 512; ofs <<= 1) {
        __syncthreads();
        int t = (tid >= ofs) ? sh[tid - ofs] : 0;
        __syncthreads();
        sh[tid] += t;
    }
    __syncthreads();
    if (tid < n) data[tid] = sh[tid] - v;
}

__global__ __launch_bounds__(256) void add_offsets_kernel(int* __restrict__ out,
                                                          const int* __restrict__ bsum, int n) {
    int i = blockIdx.x * 256 + threadIdx.x;
    if (i < n) out[i] += bsum[blockIdx.x];
}

// reorder edges by destination; fold gcn norm into per-edge weight
__global__ __launch_bounds__(256) void scatter_kernel(const int* __restrict__ src,
                                                      const int* __restrict__ dst,
                                                      const float* __restrict__ dinv,
                                                      const int* __restrict__ rowptr,
                                                      int* __restrict__ fill,
                                                      int* __restrict__ esrc,
                                                      float* __restrict__ enorm, int e_cnt) {
    int e = blockIdx.x * 256 + threadIdx.x;
    if (e >= e_cnt) return;
    int s = src[e];
    int d = dst[e];
    int pos = rowptr[d] + atomicAdd(&fill[d], 1);
    esrc[pos]  = s;
    enorm[pos] = dinv[s] * dinv[d];
}

// ------------------------------------------------------------------
// CSR propagation: xout[n] = sum_{e in in-edges(n)} w_e * xin[src_e]
// one wave per destination node; lane owns 4 floats; register
// accumulate, single non-atomic float4 store. No zero-fill needed.
// ------------------------------------------------------------------
__global__ __launch_bounds__(256) void propagate_csr_kernel(const float* __restrict__ xin,
                                                            float* __restrict__ xout,
                                                            const int* __restrict__ rowptr,
                                                            const int* __restrict__ degi,
                                                            const int* __restrict__ esrc,
                                                            const float* __restrict__ enorm) {
    int n    = blockIdx.x * 8 + (threadIdx.x >> 5);
    int lane = threadIdx.x & 31;
    if (n >= NN) return;
    int beg = rowptr[n];
    int cnt = degi[n];
    float4 acc = make_float4(0.f, 0.f, 0.f, 0.f);
    for (int j = 0; j < cnt; ++j) {
        int   s  = esrc[beg + j];
        float nv = enorm[beg + j];
        const float4 v = *(const float4*)(xin + (size_t)s * DH + lane * 4);
        acc.x += nv * v.x;
        acc.y += nv * v.y;
        acc.z += nv * v.z;
        acc.w += nv * v.w;
    }
    *(float4*)(xout + (size_t)n * DH + lane * 4) = acc;
}

// ------------------------------------------------------------------
// fused WMMA GEMM:  acc = sum_m A_m @ W_m  (+bias); optional epilogue
//   base==null : out = acc + bias
//   base!=null : out = base + coef * tanh(acc + bias)
// block = 256 threads (8 waves), 16 rows x 128 cols; wave w does the
// 16x16 tile for cols [16w,16w+16) via v_wmma_f32_16x16x32_f16.
// ------------------------------------------------------------------
template <int NMAT>
__global__ __launch_bounds__(256) void gemm_wmma_kernel(const float* __restrict__ A0,
                                                        const float* __restrict__ A1,
                                                        const float* __restrict__ A2,
                                                        const _Float16* __restrict__ W,
                                                        const float* __restrict__ bias,
                                                        const float* __restrict__ base,
                                                        float coef,
                                                        float* __restrict__ out) {
    __shared__ _Float16 As[NMAT * 16 * AS_STRIDE];

    const int tid     = threadIdx.x;
    const int rowbase = blockIdx.x * 16;

    const float* Am[3] = {A0, A1, A2};
    #pragma unroll
    for (int mat = 0; mat < NMAT; ++mat) {
        const float* A = Am[mat];
        for (int idx = tid; idx < 16 * DH; idx += 256) {
            int m = idx >> 7;
            int k = idx & 127;
            As[mat * 16 * AS_STRIDE + m * AS_STRIDE + k] =
                (_Float16)A[(size_t)(rowbase + m) * DH + k];
        }
    }
    __syncthreads();

    const int wave  = tid >> 5;
    const int lane  = tid & 31;
    const int ct    = wave;                 // column tile 0..7
    const int m     = lane & 15;            // A row within tile
    const int khalf = (lane >> 4) << 3;     // +8 k-offset for lanes 16..31

    v8f c = {};
    #pragma unroll
    for (int mat = 0; mat < NMAT; ++mat) {
        const _Float16* Wm    = W + (size_t)mat * DH * DH;
        const _Float16* asrow = &As[mat * 16 * AS_STRIDE + m * AS_STRIDE];
        #pragma unroll
        for (int kk = 0; kk < 4; ++kk) {
            // A fragment: half j of lane l = A[m][kk*32 + j + (j>=8?8:0) + (l>=16?8:0)]
            union { v16h v; unsigned u[8]; } a;
            const int kbase = kk * 32 + khalf;
            #pragma unroll
            for (int i = 0; i < 8; ++i) {
                int k0 = kbase + 2 * i + ((i >= 4) ? 8 : 0);
                a.u[i] = *(const unsigned*)(asrow + k0);   // ds_load, pair (k0,k0+1)
            }
            // B fragment: half j of lane l = W[kk*32 + j + (j>=8?8:0) + (l>=16?8:0)][ct*16 + l%16]
            v16h b;
            #pragma unroll
            for (int s = 0; s < 16; ++s)
                b[s] = Wm[(size_t)(kbase + s + ((s >= 8) ? 8 : 0)) * DH + ct * 16 + m];
            c = __builtin_amdgcn_wmma_f32_16x16x32_f16(false, a.v, false, b,
                                                       (short)0, c, false, false);
        }
    }

    // epilogue: C/D layout -> lanes 0..15: M=r, N=lane; lanes 16..31: M=r+8
    const int col    = ct * 16 + (lane & 15);
    const int rowoff = (lane >> 4) << 3;
    const float bv   = bias[col];
    #pragma unroll
    for (int r = 0; r < 8; ++r) {
        int    row = rowbase + r + rowoff;
        size_t o   = (size_t)row * DH + col;
        float  v   = c[r] + bv;
        if (base) v = base[o] + coef * tanhf(v);
        out[o] = v;
    }
}

// ------------------------------------------------------------------
// host-side orchestration
// ------------------------------------------------------------------
static inline void zero_buf(void* p, size_t n_words, hipStream_t s) {
    int n4 = (int)(n_words / 4);
    zero4_kernel<<<(n4 + 255) / 256, 256, 0, s>>>((float4*)p, n4);
}

extern "C" void kernel_launch(void* const* d_in, const int* in_sizes, int n_in,
                              void* d_out, int out_size, void* d_ws, size_t ws_size,
                              hipStream_t stream) {
    const float* x      = (const float*)d_in[0];
    const int*   ei     = (const int*)d_in[1];     // edge_index [2,E]
    const float* emb_w  = (const float*)d_in[3];
    const float* emb_b  = (const float*)d_in[4];
    const float* conv_w = (const float*)d_in[5];   // [3,128,128]
    const float* conv_b = (const float*)d_in[6];
    const float* ro_w   = (const float*)d_in[7];
    const float* ro_b   = (const float*)d_in[8];
    const int* src = ei;
    const int* dst = ei + EE;

    // workspace carve-up (256B aligned)
    char*  ws  = (char*)d_ws;
    size_t off = 0;
    auto alloc = [&](size_t bytes) -> void* {
        void* p = ws + off;
        off += (bytes + 255) & ~(size_t)255;
        return p;
    };
    int*      degi   = (int*)alloc((size_t)NN * 4);
    int*      rowptr = (int*)alloc((size_t)NN * 4);
    int*      fill   = (int*)alloc((size_t)NN * 4);
    float*    dinv   = (float*)alloc((size_t)NN * 4);
    int*      bsum   = (int*)alloc(512 * 4);
    int*      esrc   = (int*)alloc((size_t)EE * 4);
    float*    enorm  = (float*)alloc((size_t)EE * 4);
    _Float16* w16    = (_Float16*)alloc((size_t)(5 * DH * DH) * 2); // emb|conv0..2|ro
    float*    hA     = (float*)alloc((size_t)NN * DH * 4);
    float*    x1     = (float*)alloc((size_t)NN * DH * 4);
    float*    x2     = (float*)alloc((size_t)NN * DH * 4);

    float* yout = (float*)d_out;                 // [N,128] output y
    float* hmid = yout + (size_t)NN * DH;        // [N,128] output h_middle

    const _Float16* w_emb  = w16;
    const _Float16* w_conv = w16 + DH * DH;
    const _Float16* w_ro   = w16 + 4 * DH * DH;

    const int NB = (NN + 255) / 256;   // 391 scan blocks

    // ---- CSR-by-dst build + gcn norm ----
    zero_buf(degi, NN, stream);
    degi_kernel<<<(EE + 255) / 256, 256, 0, stream>>>(dst, degi, EE);
    dinv_kernel<<<NB, 256, 0, stream>>>(degi, dinv, NN);
    scan_block_kernel<<<NB, 256, 0, stream>>>(degi, rowptr, bsum, NN);
    scan_small_kernel<<<1, 512, 0, stream>>>(bsum, NB);
    add_offsets_kernel<<<NB, 256, 0, stream>>>(rowptr, bsum, NN);
    zero_buf(fill, NN, stream);
    scatter_kernel<<<(EE + 255) / 256, 256, 0, stream>>>(src, dst, dinv, rowptr,
                                                         fill, esrc, enorm, EE);

    // ---- weights -> f16 (row-major is already WMMA-B friendly) ----
    cvt_f16_kernel<<<(DH * DH + 255) / 256, 256, 0, stream>>>(emb_w, (_Float16*)w_emb, DH * DH);
    cvt_f16_kernel<<<(3 * DH * DH + 255) / 256, 256, 0, stream>>>(conv_w, (_Float16*)w_conv, 3 * DH * DH);
    cvt_f16_kernel<<<(DH * DH + 255) / 256, 256, 0, stream>>>(ro_w, (_Float16*)w_ro, DH * DH);

    const int gemm_blocks = NN / 16;   // 6250
    const int prop_blocks = NN / 8;    // 12500 (one wave per dst node)

    // ---- h = x @ emb_w + emb_b ----
    gemm_wmma_kernel<1><<<gemm_blocks, 256, 0, stream>>>(
        x, nullptr, nullptr, w_emb, emb_b, nullptr, 0.f, hA);

    // ---- 4 midpoint steps (delta_t = 4 in reference) ----
    float* hP = hA;     // current h
    float* hQ = yout;   // next-h scratch (y region, overwritten at the end)
    for (int t = 0; t < 4; ++t) {
        // conv = TAGConv(h);  h_middle = h + 0.5*eps*tanh(conv)
        propagate_csr_kernel<<<prop_blocks, 256, 0, stream>>>(hP, x1, rowptr, degi, esrc, enorm);
        propagate_csr_kernel<<<prop_blocks, 256, 0, stream>>>(x1, x2, rowptr, degi, esrc, enorm);
        gemm_wmma_kernel<3><<<gemm_blocks, 256, 0, stream>>>(
            hP, x1, x2, w_conv, conv_b, hP, 0.5f * EPSC, hmid);

        // conv_mid = TAGConv(h_middle);  h = h + eps*tanh(conv_mid)
        propagate_csr_kernel<<<prop_blocks, 256, 0, stream>>>(hmid, x1, rowptr, degi, esrc, enorm);
        propagate_csr_kernel<<<prop_blocks, 256, 0, stream>>>(x1, x2, rowptr, degi, esrc, enorm);
        gemm_wmma_kernel<3><<<gemm_blocks, 256, 0, stream>>>(
            hmid, x1, x2, w_conv, conv_b, hP, EPSC, hQ);

        float* tswap = hP; hP = hQ; hQ = tswap;
    }

    // ---- y = h_middle @ ro_w + ro_b  (overwrites the y region last) ----
    gemm_wmma_kernel<1><<<gemm_blocks, 256, 0, stream>>>(
        hmid, nullptr, nullptr, w_ro, ro_b, nullptr, 0.f, yout);

    (void)in_sizes; (void)n_in; (void)out_size; (void)ws_size;
}
